// LightRecurrentUnitBlock_60541859004653
// MI455X (gfx1250) — compile-verified, matrix operand order checked by reading, not run on hardware
//
#include <hip/hip_runtime.h>
#include <hip/hip_bf16.h>
#include <math.h>

// ---------------------------------------------------------------------------
// LightRecurrentUnitBlock for MI455X (gfx1250, wave32, WMMA 16x16x32 bf16)
//
// Pipeline (all on `stream`):
//   1. convert 9 weight matrices f32 -> bf16 (one-time, reused by all tiles)
//   2. rmsnorm: x -> xn (f32) + xn (bf16)
//   3. proj GEMM (WMMA, async-LDS double buffered): a = tanh(tanh(xn@Wi^T)),
//      f = xn@Wif^T   [layer 0]
//   4. scan layer 0 (persistent 64-WG kernel, WMMA recurrent GEMM, grid sync)
//   5. cast h1 -> bf16; proj GEMM for layer 1; scan layer 1
//   6. gate GEMMs: P1=tanh(tanh(h2@gWi^T)), P2=h2@gWif^T, P3=xn@gWhf^T
//   7. fused epilogue: out = x + xn + sigmoid(P3+gbhf+P2)*(P1-xn)
// ---------------------------------------------------------------------------

typedef __attribute__((ext_vector_type(16))) __bf16 bf16x16;
typedef __attribute__((ext_vector_type(8)))  __bf16 bf16x8;
typedef __attribute__((ext_vector_type(8)))  float  v8f;
typedef __attribute__((ext_vector_type(4)))  int    intx4;

static constexpr int    kB   = 8;
static constexpr int    kT   = 2048;
static constexpr int    kD   = 1024;
static constexpr size_t kBT  = (size_t)kB * kT;        // 16384 rows
static constexpr size_t kBTD = kBT * kD;               // 16,777,216 elems

// workspace layout (bytes); assumes ws_size >= ~403 MB
static constexpr size_t OFF_XN  = 0;                       // xn f32      [BT][D]
static constexpr size_t OFF_XNB = OFF_XN  + kBTD * 4;      // xn bf16     [BT][D]
static constexpr size_t OFF_HAB = OFF_XNB + kBTD * 2;      // h bf16      [BT][D]
static constexpr size_t OFF_A   = OFF_HAB + kBTD * 2;      // a f32       [BT][D]
static constexpr size_t OFF_F   = OFF_A   + kBTD * 4;      // f f32       [BT][D]
static constexpr size_t OFF_P3  = OFF_F   + kBTD * 4;      // p3 f32      [BT][D]
static constexpr size_t OFF_HS  = OFF_P3  + kBTD * 4;      // h_seq f32   [BT][D]
static constexpr size_t OFF_WBF = OFF_HS  + kBTD * 4;      // 9 x [D][D] bf16
static constexpr size_t OFF_HST = OFF_WBF + 9ull*kD*kD*2;  // H state bf16 [16][D]
static constexpr size_t OFF_BAR = OFF_HST + 16ull*kD*2;    // barrier counter

// CDNA5 async global->LDS path (probe via __has_builtin; safe fallback)
#if defined(__AMDGCN__) && __has_builtin(__builtin_amdgcn_global_load_async_to_lds_b128)
#define USE_ASYNC_LDS 1
#else
#define USE_ASYNC_LDS 0
#endif

__device__ __forceinline__ bf16x16 cat16(bf16x8 lo, bf16x8 hi) {
    return __builtin_shufflevector(lo, hi, 0,1,2,3,4,5,6,7,8,9,10,11,12,13,14,15);
}
__device__ __forceinline__ v8f vzero8() {
    v8f z;
#pragma unroll
    for (int i = 0; i < 8; ++i) z[i] = 0.0f;
    return z;
}
__device__ __forceinline__ float sigmoidf_(float x) {
    return 1.0f / (1.0f + __expf(-x));
}

// one 16-byte transfer global -> LDS
__device__ __forceinline__ void copy16B(const __bf16* src, __bf16* dst) {
#if USE_ASYNC_LDS
    __builtin_amdgcn_global_load_async_to_lds_b128(
        (__attribute__((address_space(1))) intx4*)(void*)src,
        (__attribute__((address_space(3))) intx4*)(void*)dst, 0, 0);
#else
    *(bf16x8*)dst = *(const bf16x8*)src;
#endif
}
__device__ __forceinline__ void wait_async_lds() {
#if USE_ASYNC_LDS
#if __has_builtin(__builtin_amdgcn_s_wait_asynccnt)
    __builtin_amdgcn_s_wait_asynccnt(0);
#else
    asm volatile("s_wait_asynccnt 0x0" ::: "memory");
#endif
#endif
}

// ---------------------------------------------------------------------------
// f32 -> bf16 elementwise convert
__global__ __launch_bounds__(256) void cvt_f32_bf16(const float* __restrict__ s,
                                                    __bf16* __restrict__ d, size_t n) {
    for (size_t i = blockIdx.x * 256ull + threadIdx.x; i < n; i += gridDim.x * 256ull)
        d[i] = (__bf16)s[i];
}

// ---------------------------------------------------------------------------
// rmsnorm: one 256-thread block per row of [BT][D]
__global__ __launch_bounds__(256) void rmsnorm_kernel(const float* __restrict__ x,
                                                      const float* __restrict__ gamma,
                                                      float* __restrict__ xn,
                                                      __bf16* __restrict__ xnb) {
    __shared__ float red[256];
    const size_t row = blockIdx.x;
    const int    tid = threadIdx.x;
    const float* xr  = x + row * kD;
    float s = 0.0f;
#pragma unroll
    for (int i = 0; i < kD / 256; ++i) {
        float v = xr[tid + i * 256];
        s += v * v;
    }
    red[tid] = s;
    __syncthreads();
    for (int off = 128; off > 0; off >>= 1) {
        if (tid < off) red[tid] += red[tid + off];
        __syncthreads();
    }
    const float nrm   = fmaxf(sqrtf(red[0]), 1e-12f);
    const float scale = 32.0f / nrm;  // sqrt(1024) = 32
#pragma unroll
    for (int i = 0; i < kD / 256; ++i) {
        const int c = tid + i * 256;
        float v = xr[c] * scale * (gamma[c] + 1.0f);
        xn[row * kD + c]  = v;
        xnb[row * kD + c] = (__bf16)v;
    }
}

// ---------------------------------------------------------------------------
// Projection GEMM: O[m][n] = epi( A[m][:] . W[n][:] ), dual-weight variant.
//   A: [M][1024] bf16 row-major, W: [1024][1024] bf16 row-major (row = out feat)
//   Workgroup: 8 waves, tile 256(M) x 64(N); each wave owns 32 rows (2 A frags)
//   so every LDS B fragment feeds two WMMAs. B chunks double-buffered in LDS,
//   streamed with GLOBAL_LOAD_ASYNC_TO_LDS_B128 when available.
//   epi == 1 -> tanh(tanh(v))  (reference applies tanh twice)
__global__ __launch_bounds__(256) void proj_gemm(const __bf16* __restrict__ A,
                                                 const __bf16* __restrict__ W0,
                                                 const __bf16* __restrict__ W1,
                                                 float* __restrict__ O0,
                                                 float* __restrict__ O1,
                                                 int epi0, int epi1) {
    __shared__ __bf16 sB[2][2][64][32];  // [buffer][weight][col][k]

    const int  tid   = threadIdx.x;
    const int  wave  = tid >> 5;
    const int  lane  = tid & 31;
    const int  lrow  = lane & 15;
    const int  khalf = (lane >> 4) * 8;
    const int  m0    = (blockIdx.x / (kD / 64)) * 256;
    const int  n0    = (blockIdx.x % (kD / 64)) * 64;
    const bool dual  = (W1 != nullptr);

    const __bf16* Arow0 = A + (size_t)(m0 + wave * 32 + lrow) * kD;
    const __bf16* Arow1 = Arow0 + 16 * (size_t)kD;
    const __bf16* W1s   = dual ? W1 : W0;

    // stage one 32-k chunk of both weight tiles (8 KB) into LDS buffer `buf`:
    // 512 x 16B transfers, 2 per thread
    auto stage = [&](int buf, int k) {
#pragma unroll
        for (int j = 0; j < 2; ++j) {
            const int idx  = tid + j * 256;      // 0..511
            const int w    = idx >> 8;           // weight select
            const int rem  = idx & 255;
            const int colL = rem >> 2;           // 0..63
            const int qo   = (rem & 3) * 8;      // 8-elem quarter within k-chunk
            const __bf16* src = ((w == 0) ? W0 : W1s) + (size_t)(n0 + colL) * kD + k + qo;
            copy16B(src, &sB[buf][w][colL][qo]);
        }
    };

    v8f acc0[2][4], acc1[2][4];
#pragma unroll
    for (int mi = 0; mi < 2; ++mi)
#pragma unroll
        for (int i = 0; i < 4; ++i) { acc0[mi][i] = vzero8(); acc1[mi][i] = vzero8(); }

    stage(0, 0);
    wait_async_lds();
    __syncthreads();

    for (int k = 0; k < kD; k += 32) {
        const int cur = (k >> 5) & 1;
        if (k + 32 < kD) stage(cur ^ 1, k + 32);  // stream next chunk during compute

        bf16x16 a0 = cat16(*(const bf16x8*)(Arow0 + k + khalf),
                           *(const bf16x8*)(Arow0 + k + khalf + 16));
        bf16x16 a1 = cat16(*(const bf16x8*)(Arow1 + k + khalf),
                           *(const bf16x8*)(Arow1 + k + khalf + 16));
        if (k + 32 < kD) {
            __builtin_prefetch(Arow0 + k + 32 + khalf, 0, 1);
            __builtin_prefetch(Arow1 + k + 32 + khalf, 0, 1);
        }

#pragma unroll
        for (int nt = 0; nt < 4; ++nt) {
            const __bf16* bp = &sB[cur][0][nt * 16 + lrow][khalf];
            bf16x16 b = cat16(*(const bf16x8*)bp, *(const bf16x8*)(bp + 16));
            acc0[0][nt] = __builtin_amdgcn_wmma_f32_16x16x32_bf16(
                false, a0, false, b, (short)0, acc0[0][nt], false, false);
            acc0[1][nt] = __builtin_amdgcn_wmma_f32_16x16x32_bf16(
                false, a1, false, b, (short)0, acc0[1][nt], false, false);
        }
        if (dual) {
#pragma unroll
            for (int nt = 0; nt < 4; ++nt) {
                const __bf16* bp = &sB[cur][1][nt * 16 + lrow][khalf];
                bf16x16 b = cat16(*(const bf16x8*)bp, *(const bf16x8*)(bp + 16));
                acc1[0][nt] = __builtin_amdgcn_wmma_f32_16x16x32_bf16(
                    false, a0, false, b, (short)0, acc1[0][nt], false, false);
                acc1[1][nt] = __builtin_amdgcn_wmma_f32_16x16x32_bf16(
                    false, a1, false, b, (short)0, acc1[1][nt], false, false);
            }
        }
        wait_async_lds();
        __syncthreads();
    }

    // epilogue: C layout lane L, vgpr r -> row r + (L>>4)*8, col L&15
#pragma unroll
    for (int mi = 0; mi < 2; ++mi) {
#pragma unroll
        for (int nt = 0; nt < 4; ++nt) {
#pragma unroll
            for (int r = 0; r < 8; ++r) {
                const int    row = m0 + wave * 32 + mi * 16 + r + (lane >> 4) * 8;
                const int    col = n0 + nt * 16 + lrow;
                const size_t idx = (size_t)row * kD + col;
                float v = acc0[mi][nt][r];
                if (epi0 == 1) v = tanhf(tanhf(v));
                O0[idx] = v;
                if (dual) {
                    float u = acc1[mi][nt][r];
                    if (epi1 == 1) u = tanhf(tanhf(u));
                    O1[idx] = u;
                }
            }
        }
    }
}

// ---------------------------------------------------------------------------
// scan init: H (padded 16 x 1024 bf16) <- h0 broadcast (rows 0..7), 0 (rows 8..15)
__global__ __launch_bounds__(256) void scan_init(__bf16* __restrict__ Hbf,
                                                 const float* __restrict__ h0,
                                                 unsigned int* __restrict__ bar) {
    const int tid = threadIdx.x;
    if (tid == 0) *bar = 0u;
    for (int i = tid; i < 16 * kD; i += 256) {
        const int r = i >> 10, n = i & (kD - 1);
        Hbf[i] = (r < kB) ? (__bf16)h0[n] : (__bf16)0.0f;
    }
}

// ---------------------------------------------------------------------------
// Persistent recurrent scan. 64 workgroups x 1 wave; WG owns 16 columns.
//   - Whf slice (16 out-features x 1024) resident in LDS for all 2048 steps
//   - H (16x1024 bf16, rows 8..15 = 0 padding) in global/L2, f32 master in regs
//   - per step: H' tile = WMMA(H, WhfSlice) over K in 4 independent chains,
//     gate + lerp in f32, write h_seq + H bf16, 64-WG grid barrier
__global__ __launch_bounds__(32) void lru_scan(const __bf16* __restrict__ Whf,
                                               const float* __restrict__ bhf,
                                               const float* __restrict__ a_in,
                                               const float* __restrict__ f_in,
                                               const float* __restrict__ h0,
                                               float* __restrict__ h_seq,
                                               __bf16* __restrict__ Hbf,
                                               unsigned int* __restrict__ bar) {
    __shared__ __bf16 sW[16][kD];  // 32 KB: this WG's Whf rows

    const int lane  = threadIdx.x;
    const int n0    = blockIdx.x * 16;
    const int lrow  = lane & 15;
    const int khalf = (lane >> 4) * 8;

    for (int i = lane; i < 16 * kD; i += 32) {
        const int r = i >> 10, kk = i & (kD - 1);
        sW[r][kk] = Whf[(size_t)(n0 + r) * kD + kk];
    }
    __syncthreads();

    const int   col  = n0 + lrow;
    const float bias = bhf[col];
    float h_reg[8];
#pragma unroll
    for (int r = 0; r < 8; ++r) h_reg[r] = h0[col];

    const unsigned nwg = gridDim.x;

    for (int t = 0; t < kT; ++t) {
        v8f acc[4];
#pragma unroll
        for (int j = 0; j < 4; ++j) acc[j] = vzero8();

        for (int k = 0; k < kD; k += 128) {
#pragma unroll
            for (int j = 0; j < 4; ++j) {
                const int ka = k + j * 32;
                bf16x16 a = cat16(*(const bf16x8*)(Hbf + lrow * kD + ka + khalf),
                                  *(const bf16x8*)(Hbf + lrow * kD + ka + khalf + 16));
                bf16x16 b = cat16(*(const bf16x8*)&sW[lrow][ka + khalf],
                                  *(const bf16x8*)&sW[lrow][ka + khalf + 16]);
                acc[j] = __builtin_amdgcn_wmma_f32_16x16x32_bf16(
                    false, a, false, b, (short)0, acc[j], false, false);
            }
        }
        v8f av = acc[0] + acc[1] + acc[2] + acc[3];

        if (lane < 16) {  // rows 0..7 (real batches) live in lanes 0..15
#pragma unroll
            for (int r = 0; r < 8; ++r) {
                const size_t idx = ((size_t)(r * kT + t)) * kD + col;
                const float  g   = sigmoidf_(av[r] + bias + f_in[idx]);
                const float  hn  = h_reg[r] + g * (a_in[idx] - h_reg[r]);
                h_reg[r]    = hn;
                h_seq[idx]  = hn;
                Hbf[r * kD + col] = (__bf16)hn;
            }
        }

        // device-wide barrier (64 WGs) so every WG sees the updated H
        __threadfence();
        if (lane == 0) {
            __hip_atomic_fetch_add(bar, 1u, __ATOMIC_ACQ_REL, __HIP_MEMORY_SCOPE_AGENT);
            const unsigned target = (unsigned)(t + 1) * nwg;
            while (__hip_atomic_load(bar, __ATOMIC_ACQUIRE, __HIP_MEMORY_SCOPE_AGENT) < target)
                __builtin_amdgcn_s_sleep(2);
        }
        __threadfence();
    }
}

// ---------------------------------------------------------------------------
// fused gate + residual epilogue:
//   g = sigmoid(P3 + gbhf + P2); y = xn + g*(P1 - xn); out = y + x
__global__ __launch_bounds__(256) void gate_residual(const float* __restrict__ x,
                                                     const float* __restrict__ xn,
                                                     const float* __restrict__ P1,
                                                     const float* __restrict__ P2,
                                                     const float* __restrict__ P3,
                                                     const float* __restrict__ gbhf,
                                                     float* __restrict__ out) {
    for (size_t i = blockIdx.x * 256ull + threadIdx.x; i < kBTD; i += gridDim.x * 256ull) {
        const int   n  = (int)(i & (kD - 1));
        const float g  = sigmoidf_(P3[i] + gbhf[n] + P2[i]);
        const float xv = xn[i];
        out[i] = xv + g * (P1[i] - xv) + x[i];
    }
}

// ---------------------------------------------------------------------------
extern "C" void kernel_launch(void* const* d_in, const int* in_sizes, int n_in,
                              void* d_out, int out_size, void* d_ws, size_t ws_size,
                              hipStream_t stream) {
    (void)in_sizes; (void)n_in; (void)out_size; (void)ws_size;

    const float* x     = (const float*)d_in[0];
    const float* gamma = (const float*)d_in[1];
    const float* gWi   = (const float*)d_in[2];
    const float* gWif  = (const float*)d_in[3];
    const float* gWhf  = (const float*)d_in[4];
    const float* gbhf  = (const float*)d_in[5];
    const float* lWi   = (const float*)d_in[6];   // [2][D][D]
    const float* lWif  = (const float*)d_in[7];
    const float* lWhf  = (const float*)d_in[8];
    const float* lbhf  = (const float*)d_in[9];   // [2][D]
    const float* h0s   = (const float*)d_in[10];  // [2][D]
    float* out = (float*)d_out;

    char* ws = (char*)d_ws;
    float*  xn   = (float*)(ws + OFF_XN);
    __bf16* xnb  = (__bf16*)(ws + OFF_XNB);
    __bf16* hab  = (__bf16*)(ws + OFF_HAB);
    float*  abuf = (float*)(ws + OFF_A);
    float*  fbuf = (float*)(ws + OFF_F);
    float*  p3   = (float*)(ws + OFF_P3);
    float*  hseq = (float*)(ws + OFF_HS);
    __bf16* wbf  = (__bf16*)(ws + OFF_WBF);
    __bf16* Hbf  = (__bf16*)(ws + OFF_HST);
    unsigned int* bar = (unsigned int*)(ws + OFF_BAR);

    const size_t DD = (size_t)kD * kD;
    __bf16* w_lWi0  = wbf + 0 * DD;
    __bf16* w_lWif0 = wbf + 1 * DD;
    __bf16* w_lWhf0 = wbf + 2 * DD;
    __bf16* w_lWi1  = wbf + 3 * DD;
    __bf16* w_lWif1 = wbf + 4 * DD;
    __bf16* w_lWhf1 = wbf + 5 * DD;
    __bf16* w_gWi   = wbf + 6 * DD;
    __bf16* w_gWif  = wbf + 7 * DD;
    __bf16* w_gWhf  = wbf + 8 * DD;

    // 1) weights -> bf16
    const int cg = 2048;
    cvt_f32_bf16<<<cg, 256, 0, stream>>>(lWi,        w_lWi0,  DD);
    cvt_f32_bf16<<<cg, 256, 0, stream>>>(lWif,       w_lWif0, DD);
    cvt_f32_bf16<<<cg, 256, 0, stream>>>(lWhf,       w_lWhf0, DD);
    cvt_f32_bf16<<<cg, 256, 0, stream>>>(lWi  + DD,  w_lWi1,  DD);
    cvt_f32_bf16<<<cg, 256, 0, stream>>>(lWif + DD,  w_lWif1, DD);
    cvt_f32_bf16<<<cg, 256, 0, stream>>>(lWhf + DD,  w_lWhf1, DD);
    cvt_f32_bf16<<<cg, 256, 0, stream>>>(gWi,        w_gWi,   DD);
    cvt_f32_bf16<<<cg, 256, 0, stream>>>(gWif,       w_gWif,  DD);
    cvt_f32_bf16<<<cg, 256, 0, stream>>>(gWhf,       w_gWhf,  DD);

    // 2) rmsnorm
    rmsnorm_kernel<<<(int)kBT, 256, 0, stream>>>(x, gamma, xn, xnb);

    const int projGrid = (int)((kBT / 256) * (kD / 64));  // 1024 WGs

    // 3-5) two LRU layers
    proj_gemm<<<projGrid, 256, 0, stream>>>(xnb, w_lWi0, w_lWif0, abuf, fbuf, 1, 0);
    scan_init<<<1, 256, 0, stream>>>(Hbf, h0s, bar);
    lru_scan<<<64, 32, 0, stream>>>(w_lWhf0, lbhf, abuf, fbuf, h0s, hseq, Hbf, bar);

    cvt_f32_bf16<<<8192, 256, 0, stream>>>(hseq, hab, kBTD);
    proj_gemm<<<projGrid, 256, 0, stream>>>(hab, w_lWi1, w_lWif1, abuf, fbuf, 1, 0);
    scan_init<<<1, 256, 0, stream>>>(Hbf, h0s + kD, bar);
    lru_scan<<<64, 32, 0, stream>>>(w_lWhf1, lbhf + kD, abuf, fbuf, h0s + kD, hseq, Hbf, bar);

    // 6) gate projections
    cvt_f32_bf16<<<8192, 256, 0, stream>>>(hseq, hab, kBTD);
    proj_gemm<<<projGrid, 256, 0, stream>>>(hab, w_gWi, w_gWif, abuf, fbuf, 1, 0);
    proj_gemm<<<projGrid, 256, 0, stream>>>(xnb, w_gWhf, nullptr, p3, nullptr, 0, 0);

    // 7) fused gate + residual
    gate_residual<<<8192, 256, 0, stream>>>(x, xn, abuf, fbuf, p3, gbhf, out);
}
